// Transformer_64768106824185
// MI455X (gfx1250) — matrix operand order repaired and checked by reading, not compiled, not run on hardware
//
#include <hip/hip_runtime.h>
#include <hip/hip_bf16.h>

typedef __attribute__((ext_vector_type(16))) _Float16 v16h;
typedef __attribute__((ext_vector_type(8)))  _Float16 v8h;
typedef __attribute__((ext_vector_type(8)))  float    v8f;

namespace {

constexpr int Bc   = 512;
constexpr int Sc   = 32;
constexpr int Dc   = 256;
constexpr int Hc   = 4;
constexpr int Ec   = 8;
constexpr int Kc   = 2;
constexpr int HIDc = 1024;
constexpr int Tc   = Bc * Sc;            // 16384 tokens
constexpr int CAPc = 2 * Tc * Kc / Ec;   // 8192 capacity per expert
constexpr int DHc  = Dc / Hc;            // 64 head dim

union V16u { v16h v; v8h h[2]; };

__device__ __forceinline__ v8f wmma_step(const V16u& a, const v16h& b, v8f c) {
  return __builtin_amdgcn_wmma_f32_16x16x32_f16(
      /*neg_a=*/false, a.v, /*neg_b=*/false, b,
      /*c_mod=*/(short)0, c, /*reuse_a=*/false, /*reuse_b=*/false);
}

// ---------------------------------------------------------------------------
// 32x32 macro-tile (2x2 WMMA tiles) per wave. A row-major [32 x K] f16,
// Bt row-major [32 x K] f16 (B transposed). Fragment layouts used
// here (contiguous K in A and B, not the cdna5_isa §7.12.2 table order):
//   A 16x32 f16: lanes 0-15 M=lane, K = g*16..g*16+15 contiguous (two 16B loads).
//   B 32x16 f16: lane n holds K = g*16..g*16+15 contiguous -> one 32B load.
// Each A/B fragment is reused by two WMMAs -> 6 loads : 4 WMMAs per K-step.
// ---------------------------------------------------------------------------
__device__ __forceinline__ void wmma_tile32(const _Float16* __restrict__ A,
                                            const _Float16* __restrict__ Bt,
                                            int K, int lane, v8f (&acc)[2][2]) {
  const int g  = (lane >> 4) & 1;
  const int mn = lane & 15;
  const _Float16* a0 = A  + (size_t)mn * K + g * 16;
  const _Float16* a1 = A  + (size_t)(mn + 16) * K + g * 16;
  const _Float16* b0 = Bt + (size_t)mn * K + g * 16;
  const _Float16* b1 = Bt + (size_t)(mn + 16) * K + g * 16;
  for (int k0 = 0; k0 < K; k0 += 32) {
    __builtin_prefetch(b0 + k0 + 128, 0, 2);   // global_prefetch_b8, next chunk
    V16u A0, A1;
    A0.h[0] = *(const v8h*)(a0 + k0);
    A0.h[1] = *(const v8h*)(a0 + k0 + 8);
    A1.h[0] = *(const v8h*)(a1 + k0);
    A1.h[1] = *(const v8h*)(a1 + k0 + 8);
    v16h B0 = *(const v16h*)(b0 + k0);
    v16h B1 = *(const v16h*)(b1 + k0);
    acc[0][0] = wmma_step(A0, B0, acc[0][0]);
    acc[0][1] = wmma_step(A0, B1, acc[0][1]);
    acc[1][0] = wmma_step(A1, B0, acc[1][0]);
    acc[1][1] = wmma_step(A1, B1, acc[1][1]);
  }
}

// ---------------------------------------------------------------------------
// Generic GEMM: C[M x N] f32 = A[M x K] f16 @ Bt[N x K]^T. 32x32/wave.
// ---------------------------------------------------------------------------
__global__ __launch_bounds__(256) void gemm_f16_f32(
    const _Float16* __restrict__ A, const _Float16* __restrict__ Bt,
    float* __restrict__ C, int M, int N, int K) {
  const int wave = (blockIdx.x * 256 + threadIdx.x) >> 5;
  const int lane = threadIdx.x & 31;
  const int nmac = N >> 5;
  const int mt = wave / nmac, nt = wave % nmac;
  if (mt >= (M >> 5)) return;
  v8f acc[2][2] = {};
  wmma_tile32(A + (size_t)(mt * 32) * K, Bt + (size_t)(nt * 32) * K, K, lane, acc);
  const int g = (lane >> 4) & 1, mn = lane & 15;
#pragma unroll
  for (int im = 0; im < 2; ++im)
#pragma unroll
    for (int in = 0; in < 2; ++in) {
      float* Crow = C + (size_t)(mt * 32 + im * 16 + g * 8) * N + nt * 32 + in * 16 + mn;
#pragma unroll
      for (int r = 0; r < 8; ++r) Crow[(size_t)r * N] = acc[im][in][r];
    }
}

// MoE GEMM1: fused bias + exact GELU, f16 output (feeds next WMMA stage).
__global__ __launch_bounds__(256) void moe_gemm1(
    const _Float16* __restrict__ A, const _Float16* __restrict__ Bt,
    const float* __restrict__ bias, _Float16* __restrict__ Hh,
    int M, int N, int K) {
  const int wave = (blockIdx.x * 256 + threadIdx.x) >> 5;
  const int lane = threadIdx.x & 31;
  const int nmac = N >> 5;
  const int mt = wave / nmac, nt = wave % nmac;
  if (mt >= (M >> 5)) return;
  v8f acc[2][2] = {};
  wmma_tile32(A + (size_t)(mt * 32) * K, Bt + (size_t)(nt * 32) * K, K, lane, acc);
  const int g = (lane >> 4) & 1, mn = lane & 15;
#pragma unroll
  for (int im = 0; im < 2; ++im)
#pragma unroll
    for (int in = 0; in < 2; ++in) {
      const int col = nt * 32 + in * 16 + mn;
      const float bv = bias[col];
      _Float16* Hrow = Hh + (size_t)(mt * 32 + im * 16 + g * 8) * N + col;
#pragma unroll
      for (int r = 0; r < 8; ++r) {
        float xv = acc[im][in][r] + bv;
        float gel = 0.5f * xv * (1.0f + erff(xv * 0.70710678118654752f));
        Hrow[(size_t)r * N] = (_Float16)gel;
      }
    }
}

// MoE GEMM2: fused bias, f32 output.
__global__ __launch_bounds__(256) void moe_gemm2(
    const _Float16* __restrict__ A, const _Float16* __restrict__ Bt,
    const float* __restrict__ bias, float* __restrict__ Y,
    int M, int N, int K) {
  const int wave = (blockIdx.x * 256 + threadIdx.x) >> 5;
  const int lane = threadIdx.x & 31;
  const int nmac = N >> 5;
  const int mt = wave / nmac, nt = wave % nmac;
  if (mt >= (M >> 5)) return;
  v8f acc[2][2] = {};
  wmma_tile32(A + (size_t)(mt * 32) * K, Bt + (size_t)(nt * 32) * K, K, lane, acc);
  const int g = (lane >> 4) & 1, mn = lane & 15;
#pragma unroll
  for (int im = 0; im < 2; ++im)
#pragma unroll
    for (int in = 0; in < 2; ++in) {
      const int col = nt * 32 + in * 16 + mn;
      const float bv = bias[col];
      float* Yrow = Y + (size_t)(mt * 32 + im * 16 + g * 8) * N + col;
#pragma unroll
      for (int r = 0; r < 8; ++r) Yrow[(size_t)r * N] = acc[im][in][r] + bv;
    }
}

// ---------------------------------------------------------------------------
// Weight prep: f32 -> f16 with transpose so GEMMs read Bt[N][K] contiguously.
// ---------------------------------------------------------------------------
__global__ void prep_weights(const float* __restrict__ qw, const float* __restrict__ kw,
                             const float* __restrict__ vw, const float* __restrict__ ow,
                             const float* __restrict__ ew1, const float* __restrict__ ew2,
                             _Float16* qT, _Float16* kT, _Float16* vT, _Float16* oT,
                             _Float16* e1T, _Float16* e2T) {
  const size_t DD = (size_t)Dc * Dc;            // 65536
  const size_t EW = (size_t)Ec * Dc * HIDc;     // 2097152
  const size_t total = 4 * DD + 2 * EW;
  for (size_t i = blockIdx.x * (size_t)blockDim.x + threadIdx.x; i < total;
       i += (size_t)gridDim.x * blockDim.x) {
    if (i < 4 * DD) {
      int wsel = (int)(i / DD);
      int r = (int)(i % DD);
      int k = r / Dc, n = r % Dc;
      const float* src = wsel == 0 ? qw : wsel == 1 ? kw : wsel == 2 ? vw : ow;
      _Float16* dst = wsel == 0 ? qT : wsel == 1 ? kT : wsel == 2 ? vT : oT;
      dst[(size_t)n * Dc + k] = (_Float16)src[r];
    } else if (i < 4 * DD + EW) {
      size_t r = i - 4 * DD;                       // e_w1 [E][D][HID]
      int e = (int)(r / ((size_t)Dc * HIDc));
      int rr = (int)(r % ((size_t)Dc * HIDc));
      int d = rr / HIDc, hd = rr % HIDc;
      e1T[((size_t)e * HIDc + hd) * Dc + d] = (_Float16)ew1[r];
    } else {
      size_t r = i - 4 * DD - EW;                  // e_w2 [E][HID][D]
      int e = (int)(r / ((size_t)HIDc * Dc));
      int rr = (int)(r % ((size_t)HIDc * Dc));
      int hd = rr / Dc, d = rr % Dc;
      e2T[((size_t)e * Dc + d) * HIDc + hd] = (_Float16)ew2[r];
    }
  }
}

// ---------------------------------------------------------------------------
// LayerNorm over D=256 (one block per token, blockDim == 256).
// ---------------------------------------------------------------------------
__global__ __launch_bounds__(256) void ln_kernel(
    const float* __restrict__ X, const float* __restrict__ w,
    const float* __restrict__ b, _Float16* __restrict__ Y16,
    float* __restrict__ Y32) {
  __shared__ float red[256];
  const int t = blockIdx.x, i = threadIdx.x;
  const float v = X[(size_t)t * Dc + i];
  red[i] = v;
  __syncthreads();
  for (int s = 128; s > 0; s >>= 1) {
    if (i < s) red[i] += red[i + s];
    __syncthreads();
  }
  const float mean = red[0] * (1.0f / Dc);
  __syncthreads();
  const float dv = v - mean;
  red[i] = dv * dv;
  __syncthreads();
  for (int s = 128; s > 0; s >>= 1) {
    if (i < s) red[i] += red[i + s];
    __syncthreads();
  }
  const float var = red[0] * (1.0f / Dc);
  const float y = dv * rsqrtf(var + 1e-6f) * w[i] + b[i];
  Y16[(size_t)t * Dc + i] = (_Float16)y;
  if (Y32) Y32[(size_t)t * Dc + i] = y;
}

// ---------------------------------------------------------------------------
// Positional projection head: ph[t][h] = pf(t) . head_w[:,h] + head_b[h].
// (pos_attn[b,h,i,j] = softmax_j(-ph[b*S+j][h]); i-terms cancel in softmax.)
// ---------------------------------------------------------------------------
__global__ void pos_kernel(const float* __restrict__ pos,
                           const float* __restrict__ p1w, const float* __restrict__ p1b,
                           const float* __restrict__ p2w, const float* __restrict__ p2b,
                           const float* __restrict__ hw, const float* __restrict__ hb,
                           float* __restrict__ ph) {
  const int t = blockIdx.x * blockDim.x + threadIdx.x;
  if (t >= Tc) return;
  const float p0 = pos[t * 3 + 0], p1 = pos[t * 3 + 1], p2 = pos[t * 3 + 2];
  float u[3];
#pragma unroll
  for (int c = 0; c < 3; ++c) {
    float s = p0 * p1w[0 * 3 + c] + p1 * p1w[1 * 3 + c] + p2 * p1w[2 * 3 + c] + p1b[c];
    u[c] = fmaxf(s, 0.0f);
  }
#pragma unroll
  for (int h = 0; h < Hc; ++h) {
    float acc = hb[h];
    for (int c = 0; c < Dc / 8; ++c) {
      float pf = u[0] * p2w[0 * 32 + c] + u[1] * p2w[1 * 32 + c] + u[2] * p2w[2 * 32 + c] + p2b[c];
      acc += pf * hw[c * Hc + h];
    }
    ph[t * Hc + h] = acc;
  }
}

// ---------------------------------------------------------------------------
// Attention per (b,h): softmax(qk^T/8) mixed with position softmax, then @ v.
// mix rows sum to (1-g)+g == 1 analytically, so renormalize is a no-op.
// ---------------------------------------------------------------------------
__global__ __launch_bounds__(256) void attn_kernel(
    const float* __restrict__ Q, const float* __restrict__ Kb,
    const float* __restrict__ V, const float* __restrict__ ph,
    const float* __restrict__ gate_p, _Float16* __restrict__ O) {
  const int bh = blockIdx.x;
  const int b = bh / Hc, h = bh % Hc;
  __shared__ float sq[Sc][DHc], sk[Sc][DHc], sv[Sc][DHc];
  __shared__ float sc[Sc][Sc];
  __shared__ float posw[Sc];

  for (int idx = threadIdx.x; idx < Sc * DHc; idx += 256) {
    const int s = idx >> 6, d = idx & 63;
    const size_t off = ((size_t)(b * Sc + s)) * Dc + h * DHc + d;
    sq[s][d] = Q[off];
    sk[s][d] = Kb[off];
    sv[s][d] = V[off];
  }
  __syncthreads();

  for (int idx = threadIdx.x; idx < Sc * Sc; idx += 256) {
    const int i = idx >> 5, j = idx & 31;
    float s = 0.0f;
    for (int d = 0; d < DHc; ++d) s += sq[i][d] * sk[j][d];
    sc[i][j] = s * 0.125f;   // 1/sqrt(64)
  }
  __syncthreads();

  const float gs = 1.0f / (1.0f + expf(-gate_p[h]));
  if (threadIdx.x < Sc) {
    float pm = -1e30f;
    for (int j = 0; j < Sc; ++j) pm = fmaxf(pm, -ph[(b * Sc + j) * Hc + h]);
    posw[threadIdx.x] = expf(-ph[(b * Sc + threadIdx.x) * Hc + h] - pm);
  }
  __syncthreads();
  if (threadIdx.x < Sc) {
    const int i = threadIdx.x;
    float m = -1e30f;
    for (int j = 0; j < Sc; ++j) m = fmaxf(m, sc[i][j]);
    float sum = 0.0f;
    for (int j = 0; j < Sc; ++j) { float e = expf(sc[i][j] - m); sc[i][j] = e; sum += e; }
    float psum = 0.0f;
    for (int j = 0; j < Sc; ++j) psum += posw[j];
    const float inv = 1.0f / sum, pinv = 1.0f / psum;
    for (int j = 0; j < Sc; ++j)
      sc[i][j] = (1.0f - gs) * sc[i][j] * inv + gs * posw[j] * pinv;
  }
  __syncthreads();

  for (int idx = threadIdx.x; idx < Sc * DHc; idx += 256) {
    const int i = idx >> 6, d = idx & 63;
    float o = 0.0f;
    for (int j = 0; j < Sc; ++j) o += sc[i][j] * sv[j][d];
    O[((size_t)(b * Sc + i)) * Dc + h * DHc + d] = (_Float16)o;
  }
}

__global__ void resid_kernel(const float* __restrict__ x, const float* __restrict__ tmp,
                             const float* __restrict__ ob, float* __restrict__ x2) {
  const int i = blockIdx.x * blockDim.x + threadIdx.x;
  if (i >= Tc * Dc) return;
  x2[i] = x[i] + tmp[i] + ob[i & (Dc - 1)];
}

__global__ void moe_init(_Float16* __restrict__ buf, int* __restrict__ counts) {
  const size_t n = (size_t)Ec * CAPc * Dc;
  for (size_t i = blockIdx.x * (size_t)blockDim.x + threadIdx.x; i < n;
       i += (size_t)gridDim.x * blockDim.x)
    buf[i] = (_Float16)0.0f;
  if (blockIdx.x == 0 && threadIdx.x < Ec) counts[threadIdx.x] = 0;
}

// Top-2 gating with per-expert atomic slot assignment (capacity 2x -> no drops).
__global__ void gate_kernel(const float* __restrict__ H2, const float* __restrict__ Wg,
                            int* __restrict__ counts, int* __restrict__ aexp,
                            int* __restrict__ aslot, float* __restrict__ awgt) {
  const int t = blockIdx.x * blockDim.x + threadIdx.x;
  if (t >= Tc) return;
  float l[Ec];
#pragma unroll
  for (int e = 0; e < Ec; ++e) {
    float s = 0.0f;
    for (int d = 0; d < Dc; ++d) s += H2[(size_t)t * Dc + d] * Wg[d * Ec + e];
    l[e] = s;
  }
  int i0 = 0;
#pragma unroll
  for (int e = 1; e < Ec; ++e) if (l[e] > l[i0]) i0 = e;
  int i1 = (i0 == 0) ? 1 : 0;
#pragma unroll
  for (int e = 0; e < Ec; ++e) if (e != i0 && l[e] > l[i1]) i1 = e;
  const float g0 = 1.0f / (1.0f + expf(l[i1] - l[i0]));
  const float g1 = 1.0f - g0;
  const int s0 = atomicAdd(&counts[i0], 1);
  const int s1 = atomicAdd(&counts[i1], 1);
  aexp[t * Kc + 0] = i0; aslot[t * Kc + 0] = (s0 < CAPc) ? s0 : -1; awgt[t * Kc + 0] = g0;
  aexp[t * Kc + 1] = i1; aslot[t * Kc + 1] = (s1 < CAPc) ? s1 : -1; awgt[t * Kc + 1] = g1;
}

// ---------------------------------------------------------------------------
// Dispatch: gather h2 rows into expert capacity buffer, staged through LDS
// with CDNA5 async global->LDS copies (ASYNCcnt) + s_wait_asynccnt.
// One wave per assignment row: 32 lanes x 16B = 512B row. Slot predicate is
// wave-uniform so there is no divergence around the async op.
// ---------------------------------------------------------------------------
__global__ __launch_bounds__(256) void dispatch_kernel(
    const _Float16* __restrict__ H16, const int* __restrict__ aexp,
    const int* __restrict__ aslot, _Float16* __restrict__ buf) {
  __shared__ _Float16 stage[8 * Dc];     // 8 waves x 256 halves (4KB), LDS base 0
  const int wave = threadIdx.x >> 5;
  const int lane = threadIdx.x & 31;
  const int a = blockIdx.x * 8 + wave;   // assignment index (grid covers Tc*Kc)
  const int t = a / Kc;
  const int s = aslot[a];

  const unsigned lds_off = (unsigned)(wave * Dc * 2 + lane * 16);
  const _Float16* src = H16 + (size_t)t * Dc + lane * 8;
  asm volatile("global_load_async_to_lds_b128 %0, %1, off"
               :: "v"(lds_off), "v"((unsigned long long)(uintptr_t)src)
               : "memory");
  asm volatile("s_wait_asynccnt 0x0" ::: "memory");

  if (s >= 0) {
    v8h* dst = (v8h*)(buf + ((size_t)aexp[a] * CAPc + s) * Dc);
    const v8h* sp = (const v8h*)&stage[wave * Dc];   // ds_load_b128
    dst[lane] = sp[lane];
  }
}

__global__ void combine_kernel(const float* __restrict__ x2, const float* __restrict__ yall,
                               const int* __restrict__ aexp, const int* __restrict__ aslot,
                               const float* __restrict__ awgt, float* __restrict__ out) {
  const int i = blockIdx.x * blockDim.x + threadIdx.x;
  if (i >= Tc * Dc) return;
  const int t = i >> 8, d = i & (Dc - 1);
  float acc = x2[i];
#pragma unroll
  for (int k = 0; k < Kc; ++k) {
    const int a = t * Kc + k;
    const int s = aslot[a];
    if (s >= 0) acc += awgt[a] * yall[((size_t)aexp[a] * CAPc + s) * Dc + d];
  }
  out[i] = acc;
}

} // namespace

extern "C" void kernel_launch(void* const* d_in, const int* in_sizes, int n_in,
                              void* d_out, int out_size, void* d_ws, size_t ws_size,
                              hipStream_t stream) {
  const float* x     = (const float*)d_in[0];
  const float* pos   = (const float*)d_in[1];
  const float* ln1w  = (const float*)d_in[2];
  const float* ln1b  = (const float*)d_in[3];
  const float* ln2w  = (const float*)d_in[4];
  const float* ln2b  = (const float*)d_in[5];
  const float* qw    = (const float*)d_in[6];
  const float* kw    = (const float*)d_in[7];
  const float* vw    = (const float*)d_in[8];
  const float* p1w   = (const float*)d_in[9];
  const float* p1b   = (const float*)d_in[10];
  const float* p2w   = (const float*)d_in[11];
  const float* p2b   = (const float*)d_in[12];
  const float* headw = (const float*)d_in[13];
  const float* headb = (const float*)d_in[14];
  const float* gatep = (const float*)d_in[15];
  const float* outw  = (const float*)d_in[16];
  const float* outb  = (const float*)d_in[17];
  const float* wgate = (const float*)d_in[18];
  const float* ew1   = (const float*)d_in[19];
  const float* eb1   = (const float*)d_in[20];
  const float* ew2   = (const float*)d_in[21];
  const float* eb2   = (const float*)d_in[22];
  float* out = (float*)d_out;

  char* ws = (char*)d_ws;
  size_t off = 0;
  auto alloc = [&](size_t bytes) -> void* {
    void* r = ws + off;
    off += (bytes + 255) & ~(size_t)255;
    return r;
  };

  _Float16* h1    = (_Float16*)alloc((size_t)Tc * Dc * 2);
  _Float16* wqT   = (_Float16*)alloc((size_t)Dc * Dc * 2);
  _Float16* wkT   = (_Float16*)alloc((size_t)Dc * Dc * 2);
  _Float16* wvT   = (_Float16*)alloc((size_t)Dc * Dc * 2);
  _Float16* woT   = (_Float16*)alloc((size_t)Dc * Dc * 2);
  float*    qbuf  = (float*)alloc((size_t)Tc * Dc * 4);
  float*    kbuf  = (float*)alloc((size_t)Tc * Dc * 4);
  float*    vbuf  = (float*)alloc((size_t)Tc * Dc * 4);
  float*    ph    = (float*)alloc((size_t)Tc * Hc * 4);
  _Float16* o16   = (_Float16*)alloc((size_t)Tc * Dc * 2);
  _Float16* h2_16 = (_Float16*)alloc((size_t)Tc * Dc * 2);
  _Float16* e1T   = (_Float16*)alloc((size_t)Ec * HIDc * Dc * 2);
  _Float16* e2T   = (_Float16*)alloc((size_t)Ec * Dc * HIDc * 2);
  _Float16* buf   = (_Float16*)alloc((size_t)Ec * CAPc * Dc * 2);
  _Float16* hh    = (_Float16*)alloc((size_t)CAPc * HIDc * 2);
  float*    yall  = (float*)alloc((size_t)Ec * CAPc * Dc * 4);
  int*      counts= (int*)alloc(Ec * 4);
  int*      aexp  = (int*)alloc((size_t)Tc * Kc * 4);
  int*      aslot = (int*)alloc((size_t)Tc * Kc * 4);
  float*    awgt  = (float*)alloc((size_t)Tc * Kc * 4);
  // q/k/v f32 buffers are dead after attention -> reuse.
  float* tmp   = qbuf;   // out-projection result
  float* x2    = kbuf;   // residual after attention
  float* h2_32 = vbuf;   // LN2 output (f32, for gating)

  // --- prep & LN1 ---
  prep_weights<<<2048, 256, 0, stream>>>(qw, kw, vw, outw, ew1, ew2,
                                         wqT, wkT, wvT, woT, e1T, e2T);
  ln_kernel<<<Tc, 256, 0, stream>>>(x, ln1w, ln1b, h1, nullptr);

  // --- QKV projections (WMMA, 32x32 macro-tiles) ---
  const int projBlocks = (Tc / 32) * (Dc / 32) / 8;  // 512
  gemm_f16_f32<<<projBlocks, 256, 0, stream>>>(h1, wqT, qbuf, Tc, Dc, Dc);
  gemm_f16_f32<<<projBlocks, 256, 0, stream>>>(h1, wkT, kbuf, Tc, Dc, Dc);
  gemm_f16_f32<<<projBlocks, 256, 0, stream>>>(h1, wvT, vbuf, Tc, Dc, Dc);

  // --- positional head + attention ---
  pos_kernel<<<(Tc + 255) / 256, 256, 0, stream>>>(pos, p1w, p1b, p2w, p2b,
                                                   headw, headb, ph);
  attn_kernel<<<Bc * Hc, 256, 0, stream>>>(qbuf, kbuf, vbuf, ph, gatep, o16);

  // --- output projection + residual ---
  gemm_f16_f32<<<projBlocks, 256, 0, stream>>>(o16, woT, tmp, Tc, Dc, Dc);
  resid_kernel<<<(Tc * Dc + 255) / 256, 256, 0, stream>>>(x, tmp, outb, x2);

  // --- LN2 + MoE routing/dispatch ---
  ln_kernel<<<Tc, 256, 0, stream>>>(x2, ln2w, ln2b, h2_16, h2_32);
  moe_init<<<4096, 256, 0, stream>>>(buf, counts);
  gate_kernel<<<(Tc + 255) / 256, 256, 0, stream>>>(h2_32, wgate, counts,
                                                    aexp, aslot, awgt);
  dispatch_kernel<<<Tc * Kc / 8, 256, 0, stream>>>(h2_16, aexp, aslot, buf);

  // --- expert FFNs (WMMA), one expert at a time to bound hh scratch ---
  const int g1Blocks = (CAPc / 32) * (HIDc / 32) / 8;  // 1024
  const int g2Blocks = (CAPc / 32) * (Dc / 32) / 8;    // 256
  for (int e = 0; e < Ec; ++e) {
    moe_gemm1<<<g1Blocks, 256, 0, stream>>>(buf + (size_t)e * CAPc * Dc,
                                            e1T + (size_t)e * HIDc * Dc,
                                            eb1 + (size_t)e * HIDc, hh,
                                            CAPc, HIDc, Dc);
    moe_gemm2<<<g2Blocks, 256, 0, stream>>>(hh,
                                            e2T + (size_t)e * Dc * HIDc,
                                            eb2 + (size_t)e * Dc,
                                            yall + (size_t)e * CAPc * Dc,
                                            CAPc, Dc, HIDc);
  }

  // --- combine: residual + weighted expert outputs ---
  combine_kernel<<<(Tc * Dc + 255) / 256, 256, 0, stream>>>(x2, yall, aexp,
                                                            aslot, awgt, out);
}
